// SpatialRelationModel_7284264534029
// MI455X (gfx1250) — compile-verified, matrix operand order checked by reading, not run on hardware
//
#include <hip/hip_runtime.h>
#include <hip/hip_bf16.h>

// ---------------------------------------------------------------------------
// SpatialRelationModel for MI455X (gfx1250, wave32, WMMA + async-to-LDS).
//
// Roofline: ~49 GFLOP total, >93% in the FFN GEMMs (82944x34 @ 34x2048 @
// 2048x34, x2 layers). Fused per-16-row-tile FFN keeps the 680MB/layer fp32
// intermediate out of HBM. v2: 8 waves per workgroup share LDS-staged,
// double-buffered W1/W2 chunks (async global->LDS, ASYNCcnt) -> 8x less L2
// weight traffic than per-wave global fragment loads, prefetch overlapped
// with v_wmma_f32_16x16x32_bf16.
// ---------------------------------------------------------------------------

typedef __attribute__((ext_vector_type(16))) __bf16 v16bf;
typedef __attribute__((ext_vector_type(8)))  float  v8f;

#define BATCH  1024
#define NTOK   81
#define DMODEL 34
#define FFDIM  2048
#define ROWS   (BATCH * NTOK)   // 82944 = 128 * 648, exactly tileable

#ifndef __has_builtin
#define __has_builtin(x) 0
#endif

#if defined(__AMDGCN__) && __has_builtin(__builtin_amdgcn_global_load_async_to_lds_b128) && \
    __has_builtin(__builtin_amdgcn_s_wait_asynccnt)
#define USE_ASYNC 1
// Parameter types per hipcc diagnostic: pointee must be the 16B int vector.
typedef int v4i_vs __attribute__((__vector_size__(4 * sizeof(int))));
#define GCAST(p) ((__attribute__((address_space(1))) v4i_vs*)(unsigned long long)(const void*)(p))
#define LCAST(p) ((__attribute__((address_space(3))) v4i_vs*)(unsigned)(unsigned long long)(const void*)(p))
#else
#define USE_ASYNC 0
#endif

union FragB { v16bf v; uint4 q[2]; };
union AccF  { v8f   v; float f[8]; };

// --------------------------- weight prep (fp32 -> bf16 padded) --------------
__global__ __launch_bounds__(256) void prep_w1_kernel(const float* __restrict__ w1,
                                                      __bf16* __restrict__ w1b) {
    int i = blockIdx.x * 256 + threadIdx.x;            // (L*FF, 64) K-padded
    if (i >= 2 * FFDIM * 64) return;
    int k = i & 63, fl = i >> 6;
    w1b[i] = (k < DMODEL) ? (__bf16)w1[fl * DMODEL + k] : (__bf16)0.0f;
}

__global__ __launch_bounds__(256) void prep_w2_kernel(const float* __restrict__ w2,
                                                      __bf16* __restrict__ w2b) {
    int i = blockIdx.x * 256 + threadIdx.x;            // (L, 48 rows, FF) N-padded
    if (i >= 2 * 48 * FFDIM) return;
    int f = i & (FFDIM - 1);
    int r = (i >> 11) % 48;
    int l = i / (48 * FFDIM);
    w2b[i] = (r < DMODEL) ? (__bf16)w2[(l * DMODEL + r) * FFDIM + f] : (__bf16)0.0f;
}

// --------------------------- conv stem + token build ------------------------
__global__ __launch_bounds__(160) void conv_tokens_kernel(
    const float* __restrict__ obs, const float* __restrict__ c1w,
    const float* __restrict__ c1b, const float* __restrict__ c2w,
    const float* __restrict__ c2b, float* __restrict__ t0) {
    __shared__ float sIn[2][9];
    __shared__ float sC1[16][9];
    __shared__ float sCell[9][16];
    int b = blockIdx.x, tid = threadIdx.x;
    if (tid < 18) ((float*)sIn)[tid] = obs[b * 18 + tid];
    __syncthreads();
    if (tid < 144) {
        int oc = tid / 9, p = tid % 9, y = p / 3, x = p % 3;
        float acc = c1b[oc];
        for (int ic = 0; ic < 2; ic++)
            for (int ky = 0; ky < 3; ky++)
                for (int kx = 0; kx < 3; kx++) {
                    int iy = y + ky - 1, ix = x + kx - 1;
                    if (iy >= 0 && iy < 3 && ix >= 0 && ix < 3)
                        acc += sIn[ic][iy * 3 + ix] * c1w[((oc * 2 + ic) * 3 + ky) * 3 + kx];
                }
        sC1[oc][p] = fmaxf(acc, 0.0f);
    }
    __syncthreads();
    if (tid < 144) {
        int oc = tid / 9, p = tid % 9, y = p / 3, x = p % 3;
        float acc = c2b[oc];
        for (int ic = 0; ic < 16; ic++)
            for (int ky = 0; ky < 3; ky++)
                for (int kx = 0; kx < 3; kx++) {
                    int iy = y + ky - 1, ix = x + kx - 1;
                    if (iy >= 0 && iy < 3 && ix >= 0 && ix < 3)
                        acc += sC1[ic][iy * 3 + ix] * c2w[((oc * 16 + ic) * 3 + ky) * 3 + kx];
                }
        sCell[p][oc] = fmaxf(acc, 0.0f);   // cell[pos][ch]
    }
    __syncthreads();
    if (tid < 81) {
        int ii = tid / 9, jj = tid % 9;
        float* out = t0 + (size_t)(b * 81 + tid) * DMODEL;
        for (int c = 0; c < 16; c++) { out[c] = sCell[ii][c]; out[16 + c] = sCell[jj][c]; }
        out[32] = (float)(ii / 3 - jj / 3) * 0.5f;
        out[33] = (float)(ii % 3 - jj % 3) * 0.5f;
    }
}

// --------------------------- attention + LN1 (one block per sample) ---------
__global__ __launch_bounds__(128) void attn_kernel(
    const float* __restrict__ tin, const float* __restrict__ wqkv,
    const float* __restrict__ bqkv, const float* __restrict__ wo,
    const float* __restrict__ bo, const float* __restrict__ ln1s,
    const float* __restrict__ ln1b, float* __restrict__ tmid,
    __bf16* __restrict__ xb) {
    __shared__ float sT[81][34];
    __shared__ float sQKV[81][102];
    __shared__ float sO[81][34];
    int b = blockIdx.x, tid = threadIdx.x;
    for (int i = tid; i < 81 * 34; i += 128) ((float*)sT)[i] = tin[(size_t)b * 81 * 34 + i];
    __syncthreads();
    for (int i = tid; i < 81 * 102; i += 128) {
        int q = i / 102, c = i % 102;
        const float* w = wqkv + c * 34;
        float acc = bqkv[c];
        for (int k = 0; k < 34; k++) acc += sT[q][k] * w[k];
        sQKV[q][c] = acc;
    }
    __syncthreads();
    const float scale = 0.2425356250363330f;   // 1/sqrt(17)
    for (int idx = tid; idx < 162; idx += 128) {
        int q = idx >> 1, h = idx & 1;
        const float* qv = &sQKV[q][h * 17];
        float mx = -3.4e38f;
        for (int kk = 0; kk < 81; kk++) {
            const float* kv = &sQKV[kk][34 + h * 17];
            float d = 0.0f;
            for (int j = 0; j < 17; j++) d += qv[j] * kv[j];
            mx = fmaxf(mx, d * scale);
        }
        float sum = 0.0f, o[17];
        for (int j = 0; j < 17; j++) o[j] = 0.0f;
        for (int kk = 0; kk < 81; kk++) {
            const float* kv = &sQKV[kk][34 + h * 17];
            float d = 0.0f;
            for (int j = 0; j < 17; j++) d += qv[j] * kv[j];
            float e = __expf(d * scale - mx);
            sum += e;
            const float* vv = &sQKV[kk][68 + h * 17];
            for (int j = 0; j < 17; j++) o[j] += e * vv[j];
        }
        float inv = 1.0f / sum;
        for (int j = 0; j < 17; j++) sO[q][h * 17 + j] = o[j] * inv;
    }
    __syncthreads();
    for (int tok = tid; tok < 81; tok += 128) {
        float x[34], mean = 0.0f;
        for (int j = 0; j < 34; j++) {
            const float* w = wo + j * 34;
            float acc = bo[j];
            for (int k = 0; k < 34; k++) acc += sO[tok][k] * w[k];
            x[j] = sT[tok][j] + acc;
            mean += x[j];
        }
        mean *= (1.0f / 34.0f);
        float var = 0.0f;
        for (int j = 0; j < 34; j++) { float d = x[j] - mean; var += d * d; }
        var *= (1.0f / 34.0f);
        float rstd = rsqrtf(var + 1e-5f);
        float*  om = tmid + (size_t)(b * 81 + tok) * DMODEL;
        __bf16* ob = xb   + (size_t)(b * 81 + tok) * 64;   // K-padded bf16 copy for FFN A-frags
        for (int j = 0; j < 34; j++) {
            float y = (x[j] - mean) * rstd * ln1s[j] + ln1b[j];
            om[j] = y;
            ob[j] = (__bf16)y;
        }
        for (int j = 34; j < 64; j++) ob[j] = (__bf16)0.0f;
    }
}

// --------------------------- fused FFN + LN2 (WMMA bf16, async staging) -----
// 256 threads = 8 wave32s, each wave owns one 16-row tile (128 rows / WG).
// Per FF chunk of 32: W1 slice (32 rows x 128B = 4KB) + W2 slice (48 rows x
// 64B = 3KB) staged once per WG into double-buffered LDS via async loads,
// consumed by all 8 waves as B-fragments. 448 b128 quads padded to 512 so
// every wave issues exactly 2 async ops -> constant s_wait_asynccnt 2 keeps
// the prefetch one chunk ahead of the WMMAs.
__device__ __forceinline__ void stage_chunk(const __bf16* __restrict__ w1base,
                                            const __bf16* __restrict__ w2base,
                                            int ffb, __bf16* sW1buf, __bf16* sW2buf,
                                            int tid) {
    const __bf16* w1c = w1base + (long)ffb * 64;       // 4KB contiguous
    for (int j = tid; j < 512; j += 256) {             // exactly 2 iters/thread
        int jj = (j < 448) ? j : (j - 448);            // pad: re-copy same data
        const __bf16* g;
        __bf16* l;
        if (jj < 256) {                                // W1: 256 quads
            g = w1c + jj * 8;
            l = sW1buf + jj * 8;
        } else {                                       // W2: 192 quads (48 rows x 4)
            int t2 = jj - 256, row = t2 >> 2, q = t2 & 3;
            g = w2base + (long)row * FFDIM + ffb + q * 8;
            l = sW2buf + row * 32 + q * 8;
        }
#if USE_ASYNC
        __builtin_amdgcn_global_load_async_to_lds_b128(GCAST(g), LCAST(l), 0, 0);
#else
        *(uint4*)l = *(const uint4*)g;
#endif
    }
}

__global__ __launch_bounds__(256) void ffn_kernel(
    const __bf16* __restrict__ xb, const float* __restrict__ xf,
    const __bf16* __restrict__ w1b, const float* __restrict__ b1,
    const __bf16* __restrict__ w2b, const float* __restrict__ b2,
    const float* __restrict__ ln2s, const float* __restrict__ ln2b,
    float* __restrict__ yout) {
    __shared__ __align__(16) __bf16 sW1[2][32 * 64];   // 2 x 4KB  W1 chunk
    __shared__ __align__(16) __bf16 sW2[2][48 * 32];   // 2 x 3KB  W2 chunk
    __shared__ __align__(16) __bf16 sH[8][16 * 32];    // per-wave H bounce (C->A relayout)
    __shared__ __align__(16) float  sE[8][16 * 48];    // per-wave epilogue tile

    int tid = threadIdx.x, wv = tid >> 5, lane = tid & 31;
    long row0 = ((long)blockIdx.x * 8 + wv) * 16;
    int  r    = lane & 15;
    int  e0   = (lane < 16) ? 0 : 8;    // first K-group element offset for frags
    int  mrow = (lane >> 4) << 3;       // C-layout row base for this lane half

    // X A-fragments: 16 rows x K=64 (padded), two K=32 frags, hoisted.
    const __bf16* xrow = xb + (row0 + r) * 64;
    FragB a0, a1;
    a0.q[0] = *(const uint4*)(xrow + e0);
    a0.q[1] = *(const uint4*)(xrow + e0 + 16);
    a1.q[0] = *(const uint4*)(xrow + 32 + e0);
    a1.q[1] = *(const uint4*)(xrow + 32 + e0 + 16);

    AccF acc[3];                         // OUT tile 16x48 (N padded 34->48), f32
    for (int t = 0; t < 3; t++)
        for (int j = 0; j < 8; j++) acc[t].f[j] = 0.0f;

    __bf16* sHw = sH[wv];

    stage_chunk(w1b, w2b, 0, sW1[0], sW2[0], tid);     // prologue: chunk 0

    for (int i = 0; i < 64; i++) {
        int ffb = i * 32;
        if (i + 1 < 64) {
            stage_chunk(w1b, w2b, ffb + 32, sW1[(i + 1) & 1], sW2[(i + 1) & 1], tid);
#if USE_ASYNC
            __builtin_amdgcn_s_wait_asynccnt(2);       // chunk i landed (in-order)
#endif
        } else {
#if USE_ASYNC
            __builtin_amdgcn_s_wait_asynccnt(0);
#endif
        }
        __syncthreads();                                // chunk i visible to all waves

        const __bf16* W1c = sW1[i & 1];
        const __bf16* W2c = sW2[i & 1];

        // GEMM1: H(16x32) = X(16x64) @ W1chunk^T, bias+relu, bf16 spill to LDS
        for (int nt = 0; nt < 2; nt++) {
            const __bf16* w1row = W1c + (nt * 16 + r) * 64;
            FragB b0, b1f;
            b0.q[0]  = *(const uint4*)(w1row + e0);
            b0.q[1]  = *(const uint4*)(w1row + e0 + 16);
            b1f.q[0] = *(const uint4*)(w1row + 32 + e0);
            b1f.q[1] = *(const uint4*)(w1row + 32 + e0 + 16);
            AccF h;
            for (int j = 0; j < 8; j++) h.f[j] = 0.0f;
            h.v = __builtin_amdgcn_wmma_f32_16x16x32_bf16(false, a0.v, false, b0.v,
                                                          (short)0, h.v, false, false);
            h.v = __builtin_amdgcn_wmma_f32_16x16x32_bf16(false, a1.v, false, b1f.v,
                                                          (short)0, h.v, false, false);
            float bias = b1[ffb + nt * 16 + r];
            for (int j = 0; j < 8; j++) {
                float hv = fmaxf(h.f[j] + bias, 0.0f);
                sHw[(mrow + j) * 32 + nt * 16 + r] = (__bf16)hv;
            }
        }
        // Re-read H as A-fragment (same-wave DS ops are in-order).
        FragB a2;
        const __bf16* hrow = sHw + r * 32;
        a2.q[0] = *(const uint4*)(hrow + e0);
        a2.q[1] = *(const uint4*)(hrow + e0 + 16);
        // GEMM2: OUT(16x48) += H(16x32) @ W2chunk^T
        for (int nt = 0; nt < 3; nt++) {
            const __bf16* w2row = W2c + (nt * 16 + r) * 32;
            FragB bf;
            bf.q[0] = *(const uint4*)(w2row + e0);
            bf.q[1] = *(const uint4*)(w2row + e0 + 16);
            acc[nt].v = __builtin_amdgcn_wmma_f32_16x16x32_bf16(false, a2.v, false, bf.v,
                                                                (short)0, acc[nt].v, false, false);
        }
        __syncthreads();                                // all reads of buf done
    }

    // Epilogue: residual + bias2 + LayerNorm via per-wave LDS tile.
    float* sEw = sE[wv];
    for (int nt = 0; nt < 3; nt++)
        for (int j = 0; j < 8; j++)
            sEw[(mrow + j) * 48 + nt * 16 + r] = acc[nt].f[j];
    if (lane < 16) {
        const float* xin = xf + (row0 + lane) * DMODEL;
        float x[34], mean = 0.0f;
        for (int c = 0; c < 34; c++) {
            x[c] = xin[c] + sEw[lane * 48 + c] + b2[c];
            mean += x[c];
        }
        mean *= (1.0f / 34.0f);
        float var = 0.0f;
        for (int c = 0; c < 34; c++) { float d = x[c] - mean; var += d * d; }
        var *= (1.0f / 34.0f);
        float rstd = rsqrtf(var + 1e-5f);
        float* yr = yout + (row0 + lane) * DMODEL;
        for (int c = 0; c < 34; c++)
            yr[c] = (x[c] - mean) * rstd * ln2s[c] + ln2b[c];
    }
}

// --------------------------- output head -----------------------------------
__global__ __launch_bounds__(64) void head_kernel(
    const float* __restrict__ tfin, const float* __restrict__ wp,
    const float* __restrict__ bp, float* __restrict__ out) {
    __shared__ float sM[34];
    int b = blockIdx.x, tid = threadIdx.x;
    if (tid < 34) {
        float acc = 0.0f;
        for (int tok = 36; tok < 45; tok++)
            acc += tfin[(size_t)(b * 81 + tok) * DMODEL + tid];
        sM[tid] = acc * (1.0f / 9.0f);
    }
    __syncthreads();
    if (tid < 4) {
        float acc = bp[tid];
        for (int k = 0; k < 34; k++) acc += sM[k] * wp[tid * 34 + k];
        out[b * 4 + tid] = acc;
    }
}

// --------------------------- launcher ---------------------------------------
extern "C" void kernel_launch(void* const* d_in, const int* in_sizes, int n_in,
                              void* d_out, int out_size, void* d_ws, size_t ws_size,
                              hipStream_t stream) {
    (void)in_sizes; (void)n_in; (void)out_size; (void)ws_size;
    const float* obs  = (const float*)d_in[0];
    const float* c1w  = (const float*)d_in[1];
    const float* c1b  = (const float*)d_in[2];
    const float* c2w  = (const float*)d_in[3];
    const float* c2b  = (const float*)d_in[4];
    const float* wqkv = (const float*)d_in[5];
    const float* bqkv = (const float*)d_in[6];
    const float* wo   = (const float*)d_in[7];
    const float* bo   = (const float*)d_in[8];
    const float* w1   = (const float*)d_in[9];
    const float* b1   = (const float*)d_in[10];
    const float* w2   = (const float*)d_in[11];
    const float* b2   = (const float*)d_in[12];
    const float* ln1s = (const float*)d_in[13];
    const float* ln1b = (const float*)d_in[14];
    const float* ln2s = (const float*)d_in[15];
    const float* ln2b = (const float*)d_in[16];
    const float* wp   = (const float*)d_in[17];
    const float* bp   = (const float*)d_in[18];

    char* ws = (char*)d_ws;
    size_t off = 0;
    float*  tA   = (float*) (ws + off); off += (size_t)ROWS * DMODEL * 4;   // 11.3 MB
    float*  tB   = (float*) (ws + off); off += (size_t)ROWS * DMODEL * 4;   // 11.3 MB
    __bf16* xbuf = (__bf16*)(ws + off); off += (size_t)ROWS * 64 * 2;       // 10.6 MB
    __bf16* w1bb = (__bf16*)(ws + off); off += (size_t)2 * FFDIM * 64 * 2;  // 512 KB
    __bf16* w2bb = (__bf16*)(ws + off); off += (size_t)2 * 48 * FFDIM * 2;  // 384 KB

    prep_w1_kernel<<<(2 * FFDIM * 64 + 255) / 256, 256, 0, stream>>>(w1, w1bb);
    prep_w2_kernel<<<(2 * 48 * FFDIM + 255) / 256, 256, 0, stream>>>(w2, w2bb);
    conv_tokens_kernel<<<BATCH, 160, 0, stream>>>(obs, c1w, c1b, c2w, c2b, tA);

    for (int l = 0; l < 2; l++) {
        attn_kernel<<<BATCH, 128, 0, stream>>>(
            tA, wqkv + (size_t)l * 102 * 34, bqkv + l * 102,
            wo + (size_t)l * 34 * 34, bo + l * 34,
            ln1s + l * 34, ln1b + l * 34, tB, xbuf);
        ffn_kernel<<<ROWS / 128, 256, 0, stream>>>(
            xbuf, tB, w1bb + (size_t)l * FFDIM * 64, b1 + l * FFDIM,
            w2bb + (size_t)l * 48 * FFDIM, b2 + l * 34,
            ln2s + l * 34, ln2b + l * 34, tA);
    }
    head_kernel<<<BATCH, 64, 0, stream>>>(tA, wp, bp, (float*)d_out);
}
